// GNN_Costume_61503931678734
// MI455X (gfx1250) — compile-verified, matrix operand order checked by reading, not run on hardware
//
#include <hip/hip_runtime.h>

#define N_NODES   100000
#define N_EDGES   1000000
#define DIM       64
#define NLAYERS   5
#define NGRAPH    128
#define NCLASS    10
#define BN_EPS    1e-5f

typedef __attribute__((ext_vector_type(16))) _Float16 v16h;
typedef __attribute__((ext_vector_type(8)))  _Float16 v8h;
typedef __attribute__((ext_vector_type(4)))  _Float16 v4h;
typedef __attribute__((ext_vector_type(8)))  float    v8f;

// ---------------------------------------------------------------------------
// Shared fragment loader (wave32, V_WMMA_F32_16X16X32_F16 operand layout).
// Works for A (rows = M of A, data row-major along K) and for B (rows = N,
// data = B^T row-major along K).  Per ISA 7.12.2: lane<16 -> row=lane,
// halves K = kb..kb+7 and kb+16..kb+23 with kb=0; lane>=16 -> kb=8.
// Both runs are 16B-contiguous -> 2 x ds_load_b128 per lane.
// ---------------------------------------------------------------------------
__device__ inline v16h frag16(const _Float16* __restrict__ base, int lane, int k0) {
    int r  = lane & 15;
    int kb = k0 + ((lane >> 4) << 3);
    v8h lo = *reinterpret_cast<const v8h*>(base + r * DIM + kb);
    v8h hi = *reinterpret_cast<const v8h*>(base + r * DIM + kb + 16);
    v16h f;
#pragma unroll
    for (int j = 0; j < 8; ++j) { f[j] = lo[j]; f[j + 8] = hi[j]; }
    return f;
}

// ---------------------------------------------------------------------------
// h[i, f] = node_table[x[i], f]
// ---------------------------------------------------------------------------
__global__ void k_init_h(const int* __restrict__ x, const float* __restrict__ table,
                         float* __restrict__ h) {
    long i = (long)blockIdx.x * blockDim.x + threadIdx.x;
    if (i < (long)N_NODES * DIM) {
        int node = (int)(i >> 6);
        int f    = (int)(i & 63);
        h[i] = table[(long)x[node] * DIM + f];
    }
}

// ---------------------------------------------------------------------------
// One wave per edge: edge_emb = attr @ ew + eb ; msg = relu(h[src]+emb);
// atomicAdd into agg[dst]. Lane handles features (lane) and (lane+32).
// ---------------------------------------------------------------------------
__global__ void __launch_bounds__(256) k_edge(const int* __restrict__ ei,
                                              const float* __restrict__ eattr,
                                              const float* __restrict__ ew,
                                              const float* __restrict__ eb,
                                              const float* __restrict__ h,
                                              float* __restrict__ agg) {
    __shared__ float sW[7 * DIM];
    __shared__ float sB[DIM];
    int t = threadIdx.x;
    for (int i = t; i < 7 * DIM; i += 256) sW[i] = ew[i];
    if (t < DIM) sB[t] = eb[t];
    __syncthreads();

    int  wid  = t >> 5;
    int  lane = t & 31;
    long e    = (long)blockIdx.x * 8 + wid;
    if (e >= N_EDGES) return;

    int src = ei[e];
    int dst = ei[(long)N_EDGES + e];
    float a[7];
#pragma unroll
    for (int j = 0; j < 7; ++j) a[j] = eattr[e * 7 + j];

#pragma unroll
    for (int half = 0; half < 2; ++half) {
        int f = lane + half * 32;
        float emb = sB[f];
#pragma unroll
        for (int j = 0; j < 7; ++j) emb += a[j] * sW[j * DIM + f];
        float m = h[(long)src * DIM + f] + emb;
        m = m > 0.f ? m : 0.f;
        atomicAdd(&agg[(long)dst * DIM + f], m);
    }
}

// ---------------------------------------------------------------------------
// Fused GIN MLP via transposed WMMA:
//   z1^T = W1^T * zin^T (relu, +b1)   ;   z^T = W2^T * z1^T (+b2)
// A = W^T tiles (staged transposed in LDS), B^T = node tile row-major.
// All fragment loads are 2x ds_load_b128; intermediate writeback is one
// packed ds_store_b128 per n-tile; final store is 2 coalesced float4s.
// ---------------------------------------------------------------------------
__global__ void __launch_bounds__(256) k_mlp(const float* __restrict__ h,
                                             const float* __restrict__ agg,
                                             const float* __restrict__ w1,
                                             const float* __restrict__ b1,
                                             const float* __restrict__ w2,
                                             const float* __restrict__ b2,
                                             float* __restrict__ z) {
    __shared__ __align__(16) _Float16 sW1t[DIM * DIM];   // W1^T : [n][k]
    __shared__ __align__(16) _Float16 sW2t[DIM * DIM];   // W2^T : [n][k]
    __shared__ __align__(16) float    sB1[DIM];
    __shared__ __align__(16) float    sB2[DIM];
    __shared__ __align__(16) _Float16 sZ[8 * 16 * DIM];  // per-wave 16x64 tile [m][k]

    int t = threadIdx.x;
    for (int i = t; i < DIM * DIM; i += 256) {
        int k = i >> 6, n = i & 63;
        sW1t[n * DIM + k] = (_Float16)w1[i];
        sW2t[n * DIM + k] = (_Float16)w2[i];
    }
    if (t < DIM) { sB1[t] = b1[t]; sB2[t] = b2[t]; }

    int  wid  = t >> 5;
    int  lane = t & 31;
    _Float16* tile = sZ + wid * (16 * DIM);
    long row0 = ((long)blockIdx.x * 8 + wid) * 16;

    // Stage zin = h + agg as f16, row-major [m][k] (OOB rows -> 0)
#pragma unroll
    for (int q = 0; q < 8; ++q) {
        int idx = lane + 32 * q;          // 256 float4s per tile
        int lin = idx * 4;
        int r   = lin >> 6;
        int col = lin & 63;
        long row = row0 + r;
        float4 v = make_float4(0.f, 0.f, 0.f, 0.f);
        if (row < N_NODES) {
            float4 a = *reinterpret_cast<const float4*>(h   + row * DIM + col);
            float4 g = *reinterpret_cast<const float4*>(agg + row * DIM + col);
            v = make_float4(a.x + g.x, a.y + g.y, a.z + g.z, a.w + g.w);
        }
        v4h p;
        p[0] = (_Float16)v.x; p[1] = (_Float16)v.y;
        p[2] = (_Float16)v.z; p[3] = (_Float16)v.w;
        *reinterpret_cast<v4h*>(tile + lin) = p;
    }
    __syncthreads();

    int mcol = lane & 15;           // node index within tile (N dim of D)
    int mb   = (lane >> 4) << 3;    // row-half offset (M dim of D)

    // ---------------- GEMM 1: z1^T = W1^T * zin^T ----------------
    v16h bz0 = frag16(tile, lane, 0);
    v16h bz1 = frag16(tile, lane, 32);
    v8f acc[4];
#pragma unroll
    for (int nt = 0; nt < 4; ++nt) {
        // bias along M (rows n): 8 consecutive values
        const float4* bp = reinterpret_cast<const float4*>(sB1 + nt * 16 + mb);
        float4 blo = bp[0], bhi = bp[1];
        v8f c;
        c[0] = blo.x; c[1] = blo.y; c[2] = blo.z; c[3] = blo.w;
        c[4] = bhi.x; c[5] = bhi.y; c[6] = bhi.z; c[7] = bhi.w;
        v16h a0 = frag16(sW1t + nt * 16 * DIM, lane, 0);
        v16h a1 = frag16(sW1t + nt * 16 * DIM, lane, 32);
        c = __builtin_amdgcn_wmma_f32_16x16x32_f16(false, a0, false, bz0, (short)0, c, false, false);
        c = __builtin_amdgcn_wmma_f32_16x16x32_f16(false, a1, false, bz1, (short)0, c, false, false);
        acc[nt] = c;
    }
    __syncthreads();

    // relu + writeback z1 row-major [m][n]: per lane 8 consecutive n
#pragma unroll
    for (int nt = 0; nt < 4; ++nt) {
        v8h p;
#pragma unroll
        for (int r = 0; r < 8; ++r) {
            float v = acc[nt][r];
            p[r] = (_Float16)(v > 0.f ? v : 0.f);
        }
        *reinterpret_cast<v8h*>(tile + mcol * DIM + nt * 16 + mb) = p;
    }
    __syncthreads();

    // ---------------- GEMM 2: z^T = W2^T * z1^T ----------------
    bz0 = frag16(tile, lane, 0);
    bz1 = frag16(tile, lane, 32);
    long row = row0 + mcol;
#pragma unroll
    for (int nt = 0; nt < 4; ++nt) {
        const float4* bp = reinterpret_cast<const float4*>(sB2 + nt * 16 + mb);
        float4 blo = bp[0], bhi = bp[1];
        v8f c;
        c[0] = blo.x; c[1] = blo.y; c[2] = blo.z; c[3] = blo.w;
        c[4] = bhi.x; c[5] = bhi.y; c[6] = bhi.z; c[7] = bhi.w;
        v16h a0 = frag16(sW2t + nt * 16 * DIM, lane, 0);
        v16h a1 = frag16(sW2t + nt * 16 * DIM, lane, 32);
        c = __builtin_amdgcn_wmma_f32_16x16x32_f16(false, a0, false, bz0, (short)0, c, false, false);
        c = __builtin_amdgcn_wmma_f32_16x16x32_f16(false, a1, false, bz1, (short)0, c, false, false);
        if (row < N_NODES) {
            float4 lo = make_float4(c[0], c[1], c[2], c[3]);
            float4 hi = make_float4(c[4], c[5], c[6], c[7]);
            float* dst = z + row * DIM + nt * 16 + mb;
            *reinterpret_cast<float4*>(dst)     = lo;
            *reinterpret_cast<float4*>(dst + 4) = hi;
        }
    }
}

// ---------------------------------------------------------------------------
// BN stats: stats[0..63]=sum, stats[64..127]=sum of squares
// ---------------------------------------------------------------------------
__global__ void __launch_bounds__(256) k_stats(const float* __restrict__ z,
                                               float* __restrict__ stats) {
    __shared__ float s[2 * DIM];
    int t = threadIdx.x;
    if (t < 2 * DIM) s[t] = 0.f;
    __syncthreads();
    long total = (long)N_NODES * DIM;
    long stride = (long)gridDim.x * blockDim.x;
    for (long i = (long)blockIdx.x * blockDim.x + t; i < total; i += stride) {
        float v = z[i];
        int f = (int)(i & 63);
        atomicAdd(&s[f], v);
        atomicAdd(&s[DIM + f], v * v);
    }
    __syncthreads();
    if (t < 2 * DIM) atomicAdd(&stats[t], s[t]);
}

// ---------------------------------------------------------------------------
// BN apply (+ optional relu), result into h ; float4 vectorized
// ---------------------------------------------------------------------------
__global__ void k_bn(const float* __restrict__ z, const float* __restrict__ stats,
                     const float* __restrict__ gamma, const float* __restrict__ beta,
                     float* __restrict__ h, int do_relu) {
    long i4 = (long)blockIdx.x * blockDim.x + threadIdx.x;
    if (i4 < (long)N_NODES * DIM / 4) {
        long base = i4 * 4;
        int f0 = (int)(base & 63);
        const float inv_n = 1.f / (float)N_NODES;
        float4 v = *reinterpret_cast<const float4*>(z + base);
        float o[4] = {v.x, v.y, v.z, v.w};
#pragma unroll
        for (int j = 0; j < 4; ++j) {
            int f = f0 + j;
            float mu  = stats[f] * inv_n;
            float var = stats[DIM + f] * inv_n - mu * mu;
            float r = (o[j] - mu) * rsqrtf(var + BN_EPS) * gamma[f] + beta[f];
            if (do_relu) r = r > 0.f ? r : 0.f;
            o[j] = r;
        }
        *reinterpret_cast<float4*>(h + base) = make_float4(o[0], o[1], o[2], o[3]);
    }
}

// ---------------------------------------------------------------------------
// Pooling
// ---------------------------------------------------------------------------
__global__ void k_pool(const float* __restrict__ h, const int* __restrict__ batch,
                       float* __restrict__ pool) {
    long i = (long)blockIdx.x * blockDim.x + threadIdx.x;
    if (i < (long)N_NODES * DIM) {
        int node = (int)(i >> 6);
        int f    = (int)(i & 63);
        atomicAdd(&pool[(long)batch[node] * DIM + f], h[i]);
    }
}

__global__ void k_cnt(const int* __restrict__ batch, float* __restrict__ cnt) {
    int i = blockIdx.x * blockDim.x + threadIdx.x;
    if (i < N_NODES) atomicAdd(&cnt[batch[i]], 1.f);
}

__global__ void k_head(const float* __restrict__ pool, const float* __restrict__ cnt,
                       const float* __restrict__ pw, const float* __restrict__ pb,
                       float* __restrict__ out) {
    int i = blockIdx.x * blockDim.x + threadIdx.x;
    if (i < NGRAPH * NCLASS) {
        int g = i / NCLASS, c = i % NCLASS;
        float inv = 1.f / fmaxf(cnt[g], 1.f);
        float acc = pb[c];
#pragma unroll
        for (int f = 0; f < DIM; ++f) acc += pool[g * DIM + f] * inv * pw[f * NCLASS + c];
        out[i] = acc;
    }
}

// ---------------------------------------------------------------------------
extern "C" void kernel_launch(void* const* d_in, const int* in_sizes, int n_in,
                              void* d_out, int out_size, void* d_ws, size_t ws_size,
                              hipStream_t stream) {
    const int*   x     = (const int*)d_in[0];
    const int*   ei    = (const int*)d_in[1];
    const float* eattr = (const float*)d_in[2];
    const int*   batch = (const int*)d_in[3];
    const float* table = (const float*)d_in[4];
    const float* ew    = (const float*)d_in[5];
    const float* eb    = (const float*)d_in[6];
    const float* w1    = (const float*)d_in[7];
    const float* b1    = (const float*)d_in[8];
    const float* w2    = (const float*)d_in[9];
    const float* b2    = (const float*)d_in[10];
    const float* gamma = (const float*)d_in[11];
    const float* beta  = (const float*)d_in[12];
    const float* pw    = (const float*)d_in[13];
    const float* pb    = (const float*)d_in[14];
    float* out = (float*)d_out;

    float* ws    = (float*)d_ws;
    float* h     = ws;                                  // N*DIM
    float* agg   = h   + (size_t)N_NODES * DIM;         // N*DIM
    float* z     = agg + (size_t)N_NODES * DIM;         // N*DIM
    float* stats = z   + (size_t)N_NODES * DIM;         // 2*DIM
    float* pool  = stats + 2 * DIM;                     // NGRAPH*DIM
    float* cnt   = pool  + NGRAPH * DIM;                // NGRAPH

    const int ELEM_BLOCKS = (N_NODES * DIM + 255) / 256;

    k_init_h<<<ELEM_BLOCKS, 256, 0, stream>>>(x, table, h);

    for (int l = 0; l < NLAYERS; ++l) {
        hipMemsetAsync(agg, 0, sizeof(float) * (size_t)N_NODES * DIM, stream);
        k_edge<<<(N_EDGES + 7) / 8, 256, 0, stream>>>(
            ei, eattr, ew + (size_t)l * 7 * DIM, eb + (size_t)l * DIM, h, agg);
        k_mlp<<<(N_NODES + 127) / 128, 256, 0, stream>>>(
            h, agg, w1 + (size_t)l * DIM * DIM, b1 + (size_t)l * DIM,
            w2 + (size_t)l * DIM * DIM, b2 + (size_t)l * DIM, z);
        hipMemsetAsync(stats, 0, sizeof(float) * 2 * DIM, stream);
        k_stats<<<480, 256, 0, stream>>>(z, stats);
        k_bn<<<(N_NODES * DIM / 4 + 255) / 256, 256, 0, stream>>>(
            z, stats, gamma + (size_t)l * DIM, beta + (size_t)l * DIM, h,
            (l != NLAYERS - 1) ? 1 : 0);
    }

    hipMemsetAsync(pool, 0, sizeof(float) * (NGRAPH * DIM + NGRAPH), stream);
    k_pool<<<ELEM_BLOCKS, 256, 0, stream>>>(h, batch, pool);
    k_cnt<<<(N_NODES + 255) / 256, 256, 0, stream>>>(batch, cnt);
    k_head<<<(NGRAPH * NCLASS + 255) / 256, 256, 0, stream>>>(pool, cnt, pw, pb, out);
}
